// Single_Layer_Hybrid_Attention_11587821764877
// MI455X (gfx1250) — matrix operand order repaired and checked by reading, not compiled, not run on hardware
//
#include <hip/hip_runtime.h>

typedef __bf16 bf16;
typedef __attribute__((ext_vector_type(16))) __bf16 v16bf;
typedef __attribute__((ext_vector_type(8)))  __bf16 v8bf;
typedef __attribute__((ext_vector_type(8)))  float   v8f;

#define B_BATCH 240
#define PAD     95
#define NTOK    13320
#define DM      512
#define DI      2048
#define M_VALID (B_BATCH * PAD)      // 22800
#define M2      22848                // 357 * 64, padded row count
#define LDA     40                   // LDS tile row stride in elems (80B)

#define AS1 __attribute__((address_space(1)))
#define AS3 __attribute__((address_space(3)))

typedef int v4i_vs __attribute__((vector_size(16)));   // builtin's b128 unit

// ---------------------------------------------------------------------------
// Feature ladder (device-pass only; host pass sees plain C++ fallbacks)
// ---------------------------------------------------------------------------
#if defined(__gfx1250__)
# if __has_builtin(__builtin_amdgcn_global_load_async_to_lds_b128)
#  define HAVE_ASYNC_CP 1
# else
#  define HAVE_ASYNC_CP 0
# endif
# if __has_builtin(__builtin_amdgcn_tensor_load_to_lds) && \
     __has_builtin(__builtin_amdgcn_s_wait_tensorcnt)
#  define HAVE_TDM 1
# else
#  define HAVE_TDM 0
# endif
#else
# define HAVE_ASYNC_CP 0
# define HAVE_TDM 0
#endif

#if HAVE_ASYNC_CP
# if __has_builtin(__builtin_amdgcn_s_wait_asynccnt)
#  define WAIT_ASYNC(n) __builtin_amdgcn_s_wait_asynccnt(n)
# else
#  define WAIT_ASYNC(n) asm volatile("s_wait_asynccnt %0" :: "i"(n) : "memory")
# endif
#else
# define WAIT_ASYNC(n) ((void)0)
#endif

#if HAVE_TDM
# define WAIT_TENSOR(n) __builtin_amdgcn_s_wait_tensorcnt(n)
# define ASYNC_PER_BUF 1            // only A tile goes through ASYNCcnt
#else
# define WAIT_TENSOR(n) ((void)0)
# define ASYNC_PER_BUF 3            // A (1) + B (2) async ops per wave
#endif

// One 16-byte global->LDS transfer (async if available, else sync staging)
__device__ __forceinline__ void cp_b128(bf16* ldst, const bf16* gsrc) {
#if HAVE_ASYNC_CP
  __builtin_amdgcn_global_load_async_to_lds_b128(
      (AS1 v4i_vs*)(const AS1 void*)gsrc,
      (AS3 v4i_vs*)(AS3 void*)ldst, 0, 0);
#else
  *(v8bf*)ldst = *(const v8bf*)gsrc;
#endif
}

#if HAVE_TDM
typedef __attribute__((ext_vector_type(4))) unsigned int u32x4;
typedef __attribute__((ext_vector_type(4))) int          i32x4;
typedef __attribute__((ext_vector_type(8))) int          i32x8;

// TDM: load a 128x32 bf16 tile (rows of 64B) from a [rows, ldk] bf16 tensor
// into LDS; pad_enable inserts 16B after every 64B row -> LDS stride 80B (LDA).
__device__ __forceinline__ void tdm_load_tile(bf16* lds_dst, const bf16* gsrc,
                                              unsigned ldk) {
  unsigned lds_off = (unsigned)(unsigned long long)(const AS3 void*)lds_dst;
  unsigned long long ga = (unsigned long long)gsrc;
  u32x4 g0;
  g0.x = 1u;                                             // count=1 (user D#)
  g0.y = lds_off;                                        // lds_addr
  g0.z = (unsigned)ga;                                   // global_addr[31:0]
  g0.w = (unsigned)((ga >> 32) & 0x01FFFFFFull) | (2u << 30);  // addr hi | type=2
  i32x8 g1;
  g1[0] = (int)((1u << 16) |          // data_size = 2 bytes
                (1u << 20) |          // pad_enable
                (3u << 22) |          // pad_interval: 16 DWORDs (one 64B row)
                (3u << 25));          // pad_amount:   4 DWORDs (16B)
  g1[1] = (int)((ldk & 0xFFFFu) << 16);                        // tensor_dim0 lo
  g1[2] = (int)(((ldk >> 16) & 0xFFFFu) | (128u << 16));       // d0 hi | tensor_dim1=128
  g1[3] = (int)(32u << 16);                                    // tile_dim0 = 32
  g1[4] = 128;                                                 // tile_dim1 = 128
  g1[5] = (int)ldk;                                            // tensor_dim0_stride lo
  g1[6] = 0;
  g1[7] = 0;
  i32x4 z4 = {0, 0, 0, 0};
#if __clang_major__ >= 23
  i32x8 z8 = {0, 0, 0, 0, 0, 0, 0, 0};
  __builtin_amdgcn_tensor_load_to_lds(g0, g1, z4, z4, z8, 0);
#else
  __builtin_amdgcn_tensor_load_to_lds(g0, g1, z4, z4, 0);
#endif
}
#endif

// ---------------------------------------------------------------------------
// WMMA helpers
// ---------------------------------------------------------------------------
__device__ __forceinline__ v8f wmma_bf16(v16bf a, v16bf b, v8f c) {
  return __builtin_amdgcn_wmma_f32_16x16x32_bf16(
      /*neg_a=*/false, a, /*neg_b=*/false, b,
      /*c_mod=*/(short)0, c, /*reuse_a=*/false, /*reuse_b=*/false);
}

// A-matrix 16x32 bf16 fragment: chunks at k0+hf*8 and k0+16+hf*8
__device__ __forceinline__ v16bf frag_a(const bf16* base, int ld, int row, int k0, int hf) {
  const bf16* p = base + (size_t)row * ld + k0 + hf * 8;
  union { v16bf v; v8bf h[2]; } u;
  u.h[0] = *(const v8bf*)(p);
  u.h[1] = *(const v8bf*)(p + 16);
  return u.v;
}

// B-matrix 32x16 bf16 fragment: chunks at k0+hf*8 and k0+16+hf*8
__device__ __forceinline__ v16bf frag_b(const bf16* base, int ld, int row, int k0, int hf) {
  const bf16* p = base + (size_t)row * ld + k0 + hf * 8;
  union { v16bf v; v8bf h[2]; } u;
  u.h[0] = *(const v8bf*)(p);
  u.h[1] = *(const v8bf*)(p + 16);
  return u.v;
}

// ---------------------------------------------------------------------------
// Utility kernels
// ---------------------------------------------------------------------------
__global__ void prefix_kernel(const int* __restrict__ num_objs, int* __restrict__ offs) {
  if (threadIdx.x == 0) {
    int a = 0;
    for (int i = 0; i < B_BATCH; ++i) { offs[i] = a; a += num_objs[i]; }
    offs[B_BATCH] = a;
  }
}

__global__ __launch_bounds__(256)
void pad_kernel(const float* __restrict__ vis, const float* __restrict__ txt,
                const int* __restrict__ num_objs, const int* __restrict__ offs,
                float* __restrict__ vis_f, float* __restrict__ txt_f,
                bf16* __restrict__ vis_b, bf16* __restrict__ txt_b) {
  int row = blockIdx.x;                 // 0 .. M2-1
  int bb = row / PAD, t = row % PAD;
  bool inb = row < M_VALID;
  int len = inb ? num_objs[bb] : 0;
  bool valid = inb && (t < len);
  int src = valid ? (offs[bb] + t) : 0;
  for (int j = 0; j < 2; ++j) {
    int c = threadIdx.x + j * 256;
    float v = valid ? vis[(size_t)src * DM + c] : 0.f;
    float w = valid ? txt[(size_t)src * DM + c] : 0.f;
    size_t idx = (size_t)row * DM + c;
    vis_f[idx] = v;  txt_f[idx] = w;
    vis_b[idx] = (bf16)v;  txt_b[idx] = (bf16)w;
  }
}

// src f32 [K, N] row-major  ->  dst bf16 [N, K] row-major (transposed cast)
__global__ __launch_bounds__(256)
void cvtw_kernel(const float* __restrict__ src, bf16* __restrict__ dst, int Kd, int Nd) {
  int idx = blockIdx.x * 256 + threadIdx.x;
  if (idx < Kd * Nd) {
    int k = idx / Nd, n = idx % Nd;
    dst[(size_t)n * Kd + k] = (bf16)src[idx];
  }
}

__global__ __launch_bounds__(256)
void unpad_kernel(const float* __restrict__ sv, const float* __restrict__ st,
                  const int* __restrict__ offs, float* __restrict__ out) {
  int i = blockIdx.x;                   // packed token index
  __shared__ int sb;
  if (threadIdx.x == 0) {
    int lo = 0, hi = B_BATCH;
    while (hi - lo > 1) { int mid = (lo + hi) >> 1; if (offs[mid] <= i) lo = mid; else hi = mid; }
    sb = lo;
  }
  __syncthreads();
  int bb = sb;
  int t = i - offs[bb];
  size_t srow = (size_t)(bb * PAD + t) * DM;
  for (int j = threadIdx.x; j < DM; j += 256) {
    out[(size_t)i * DM + j]          = sv[srow + j];
    out[(size_t)(NTOK + i) * DM + j] = st[srow + j];
  }
}

// ---------------------------------------------------------------------------
// GEMM:  C[M2, N] = A_bf16[M2, K] @ Bt_bf16[N, K]^T  (+ bias, epilogues)
//   EP 0: bf16 out, +bias    EP 1: bf16 out, +bias, ReLU
//   EP 2: f32 out, +bias, +residual
// Block = 256 threads (8 waves, 2x4), block tile 64x128, wave tile 32x32.
// Double-buffered LDS staging: B tile via TDM (descriptor with pad_enable),
// A tile via async global->LDS b128, sync fallback otherwise.
// ---------------------------------------------------------------------------
template <int EP>
__global__ __launch_bounds__(256)
void gemm_wmma(const bf16* __restrict__ A, const bf16* __restrict__ Bt,
               const float* __restrict__ bias, const float* __restrict__ Res,
               float* __restrict__ Of, bf16* __restrict__ Ob,
               int Ncols, int K) {
  __shared__ __align__(16) bf16 As[2][64 * LDA];
  __shared__ __align__(16) bf16 Bs[2][128 * LDA];

  const int tid  = threadIdx.x;
  const int lane = tid & 31;
  const int wave = tid >> 5;
  const int wm = wave >> 2, wn = wave & 3;
  const int hf = lane >> 4, l16 = lane & 15;
  const int mblk = blockIdx.x * 64;
  const int nblk = blockIdx.y * 128;
  const int am0 = wm * 32;
  const int bn0 = wn * 32;
  const int arow = tid >> 2, aseg = tid & 3;   // A tile staging coords

  auto stage = [&](int step, int buf) {
    const int k0 = step * 32;
    // A tile 64x32: one 16B transfer per thread
    cp_b128(&As[buf][arow * LDA + aseg * 8],
            A + (size_t)(mblk + arow) * K + k0 + aseg * 8);
#if HAVE_TDM
    if (wave == 0)
      tdm_load_tile(&Bs[buf][0], Bt + (size_t)nblk * K + k0, (unsigned)K);
#else
    // B tile 128x32: two 16B transfers per thread
    for (int j = 0; j < 2; ++j) {
      int i = tid + j * 256;
      int br = i >> 2, bs = i & 3;
      cp_b128(&Bs[buf][br * LDA + bs * 8],
              Bt + (size_t)(nblk + br) * K + k0 + bs * 8);
    }
#endif
  };

  v8f acc00 = {}, acc01 = {}, acc10 = {}, acc11 = {};
  const int nsteps = K >> 5;

  stage(0, 0);
  for (int it = 0; it < nsteps; ++it) {
    const int cur = it & 1;
    const bool more = (it + 1) < nsteps;
    if (more) stage(it + 1, cur ^ 1);
    if (more) { WAIT_ASYNC(ASYNC_PER_BUF); } else { WAIT_ASYNC(0); }
#if HAVE_TDM
    if (wave == 0) { if (more) WAIT_TENSOR(1); else WAIT_TENSOR(0); }
#endif
    __syncthreads();
    const bf16* Ac = As[cur];
    const bf16* Bc = Bs[cur];
    v16bf a0 = frag_a(Ac, LDA, am0 + l16,      0, hf);
    v16bf a1 = frag_a(Ac, LDA, am0 + 16 + l16, 0, hf);
    v16bf b0 = frag_b(Bc, LDA, bn0 + l16,      0, hf);
    v16bf b1 = frag_b(Bc, LDA, bn0 + 16 + l16, 0, hf);
    acc00 = wmma_bf16(a0, b0, acc00);
    acc01 = wmma_bf16(a0, b1, acc01);
    acc10 = wmma_bf16(a1, b0, acc10);
    acc11 = wmma_bf16(a1, b1, acc11);
    __syncthreads();
  }

  auto epi = [&](v8f a, int im, int in) {
    for (int r = 0; r < 8; ++r) {
      int row = mblk + am0 + im * 16 + r + 8 * hf;
      int col = nblk + bn0 + in * 16 + l16;
      size_t idx = (size_t)row * Ncols + col;
      float v = a[r] + bias[col];
      if constexpr (EP == 1) v = fmaxf(v, 0.f);
      if constexpr (EP == 2) Of[idx] = v + Res[idx];
      else                   Ob[idx] = (bf16)v;
    }
  };
  epi(acc00, 0, 0); epi(acc01, 0, 1); epi(acc10, 1, 0); epi(acc11, 1, 1);
}

// ---------------------------------------------------------------------------
// Attention: one block per (batch, head). 192 threads = 6 waves; wave w owns
// row tile w (16 rows of the 96-padded sequence).
// ---------------------------------------------------------------------------
__global__ __launch_bounds__(192)
void attn_wmma(const bf16* __restrict__ Q, const bf16* __restrict__ Kmat,
               const bf16* __restrict__ V, bf16* __restrict__ Ctx,
               const int* __restrict__ num_objs) {
  __shared__ __align__(16) bf16 Vt[64 * 96];   // [d][t], V transposed
  __shared__ __align__(16) bf16 Pp[96 * 96];   // softmax probs, bf16

  int bb = blockIdx.x, h = blockIdx.y;
  int len = num_objs[bb];
  int tid = threadIdx.x;
  int lane = tid & 31, wave = tid >> 5;
  int hf = lane >> 4, l16 = lane & 15;
  int rowbase = bb * PAD;

  const bf16* Qh = Q    + (size_t)rowbase * DM + h * 64;
  const bf16* Kh = Kmat + (size_t)rowbase * DM + h * 64;
  const bf16* Vh = V    + (size_t)rowbase * DM + h * 64;

  for (int i = tid; i < 64 * 96; i += 192) {
    int t = i >> 6, d = i & 63;
    Vt[d * 96 + t] = Vh[(size_t)t * DM + d];
  }
  __syncthreads();

  int m0 = wave * 16;

  // Scores: S[16 x 96] = Q_tile @ K^T  (K rows are directly the B operand)
  v8f s[6] = {};
  for (int k0 = 0; k0 < 64; k0 += 32) {
    v16bf qa = frag_a(Qh, DM, m0 + l16, k0, hf);
    for (int n = 0; n < 6; ++n) {
      v16bf kb = frag_b(Kh, DM, n * 16 + l16, k0, hf);
      s[n] = wmma_bf16(qa, kb, s[n]);
    }
  }

  // Scale, mask padded keys, row-wise softmax (rows live in 16-lane groups)
  float rmax[8], rsum[8], rinv[8];
  for (int r = 0; r < 8; ++r) rmax[r] = -3.4e38f;
  for (int n = 0; n < 6; ++n) {
    int col = n * 16 + l16;
    for (int r = 0; r < 8; ++r) {
      float v = s[n][r] * 0.125f;            // 1/sqrt(64)
      if (col >= len) v = -1e30f;
      s[n][r] = v;
      rmax[r] = fmaxf(rmax[r], v);
    }
  }
  for (int off = 8; off; off >>= 1)
    for (int r = 0; r < 8; ++r)
      rmax[r] = fmaxf(rmax[r], __shfl_xor(rmax[r], off, 32));
  for (int r = 0; r < 8; ++r) rsum[r] = 0.f;
  for (int n = 0; n < 6; ++n)
    for (int r = 0; r < 8; ++r) {
      float p = __expf(s[n][r] - rmax[r]);
      s[n][r] = p;
      rsum[r] += p;
    }
  for (int off = 8; off; off >>= 1)
    for (int r = 0; r < 8; ++r)
      rsum[r] += __shfl_xor(rsum[r], off, 32);
  for (int r = 0; r < 8; ++r) rinv[r] = 1.f / rsum[r];

  for (int n = 0; n < 6; ++n)
    for (int r = 0; r < 8; ++r)
      Pp[(m0 + r + 8 * hf) * 96 + n * 16 + l16] = (bf16)(s[n][r] * rinv[r]);
  __syncthreads();

  // Context: C[16 x 64] = P[16 x 96] @ V[96 x 64] using transposed V in LDS
  v8f c[4] = {};
  for (int k0 = 0; k0 < 96; k0 += 32) {
    v16bf pa = frag_a((const bf16*)Pp, 96, m0 + l16, k0, hf);
    for (int d = 0; d < 4; ++d) {
      v16bf vb = frag_b((const bf16*)Vt, 96, d * 16 + l16, k0, hf);
      c[d] = wmma_bf16(pa, vb, c[d]);
    }
  }
  for (int d = 0; d < 4; ++d)
    for (int r = 0; r < 8; ++r) {
      int m = m0 + r + 8 * hf;
      if (m < PAD)                          // row 95 belongs to next batch
        Ctx[(size_t)(rowbase + m) * DM + h * 64 + d * 16 + l16] = (bf16)c[d][r];
    }
}

// ---------------------------------------------------------------------------
// LayerNorm (+mask, optional bf16 copy, optional accumulate into dst)
// ---------------------------------------------------------------------------
__global__ __launch_bounds__(256)
void ln_kernel(const float* __restrict__ X, const float* __restrict__ g,
               const float* __restrict__ be, const int* __restrict__ num_objs,
               float* __restrict__ Of, bf16* __restrict__ Ob, int accumulate) {
  int row = blockIdx.x;
  int tid = threadIdx.x;
  size_t base = (size_t)row * DM;
  float x0 = X[base + tid], x1 = X[base + 256 + tid];
  float s = x0 + x1, s2 = x0 * x0 + x1 * x1;
  for (int off = 16; off; off >>= 1) {
    s  += __shfl_xor(s,  off, 32);
    s2 += __shfl_xor(s2, off, 32);
  }
  __shared__ float rs[8], rs2[8];
  __shared__ float mean_s, rstd_s;
  int wv = tid >> 5, lid = tid & 31;
  if (lid == 0) { rs[wv] = s; rs2[wv] = s2; }
  __syncthreads();
  if (tid == 0) {
    float a = 0.f, b2 = 0.f;
    for (int i = 0; i < 8; ++i) { a += rs[i]; b2 += rs2[i]; }
    float mean = a * (1.f / DM);
    float var = b2 * (1.f / DM) - mean * mean;
    mean_s = mean;
    rstd_s = rsqrtf(var + 1e-5f);
  }
  __syncthreads();
  int bbi = row / PAD, t = row % PAD;
  int len = (row < M_VALID) ? num_objs[bbi] : 0;
  float mk = (row < M_VALID && t < len) ? 1.f : 0.f;
  float mean = mean_s, rstd = rstd_s;
  float xv[2] = { x0, x1 };
  for (int j = 0; j < 2; ++j) {
    int cc = tid + j * 256;
    float y = ((xv[j] - mean) * rstd * g[cc] + be[cc]) * mk;
    if (accumulate) Of[base + cc] += y; else Of[base + cc] = y;
    if (Ob) Ob[base + cc] = (bf16)y;
  }
}

// ---------------------------------------------------------------------------
// Host launcher
// ---------------------------------------------------------------------------
extern "C" void kernel_launch(void* const* d_in, const int* in_sizes, int n_in,
                              void* d_out, int out_size, void* d_ws, size_t ws_size,
                              hipStream_t stream) {
  const float* visual = (const float*)d_in[0];
  const float* text   = (const float*)d_in[1];
  const float* Wq = (const float*)d_in[2];   const float* bq = (const float*)d_in[3];
  const float* Wk = (const float*)d_in[4];   const float* bk = (const float*)d_in[5];
  const float* Wv = (const float*)d_in[6];   const float* bv = (const float*)d_in[7];
  const float* Wo = (const float*)d_in[8];   const float* bo = (const float*)d_in[9];
  const float* g1 = (const float*)d_in[10];  const float* be1 = (const float*)d_in[11];
  const float* W1 = (const float*)d_in[12];  const float* b1 = (const float*)d_in[13];
  const float* W2 = (const float*)d_in[14];  const float* b2 = (const float*)d_in[15];
  const float* g2 = (const float*)d_in[16];  const float* be2 = (const float*)d_in[17];
  const int* num_objs = (const int*)d_in[18];
  float* out = (float*)d_out;

  char* ws = (char*)d_ws;
  size_t off = 0;
  auto take = [&](size_t bytes) -> char* {
    char* p = ws + off;
    off += (bytes + 255) & ~(size_t)255;
    return p;
  };

  const size_t S512f = (size_t)M2 * DM * sizeof(float);
  const size_t S512b = (size_t)M2 * DM * sizeof(bf16);
  const size_t S2048b = (size_t)M2 * DI * sizeof(bf16);

  int*   offs    = (int*)  take(sizeof(int) * (B_BATCH + 1));
  float* vis_f   = (float*)take(S512f);
  float* txt_f   = (float*)take(S512f);
  bf16*  vis_b   = (bf16*) take(S512b);
  bf16*  txt_b   = (bf16*) take(S512b);
  bf16*  Qb      = (bf16*) take(S512b);
  bf16*  Kb      = (bf16*) take(S512b);
  bf16*  Vb      = (bf16*) take(S512b);
  bf16*  Ctx     = (bf16*) take(S512b);
  float* Tf      = (float*)take(S512f);
  float* O1f     = (float*)take(S512f);
  bf16*  O1b     = (bf16*) take(S512b);
  bf16*  Hb      = (bf16*) take(S2048b);
  float* sum_vis = (float*)take(S512f);
  float* sum_txt = (float*)take(S512f);
  bf16*  WqT     = (bf16*) take((size_t)4 * DM * DM * sizeof(bf16));
  bf16*  WkT     = (bf16*) take((size_t)4 * DM * DM * sizeof(bf16));
  bf16*  WvT     = (bf16*) take((size_t)4 * DM * DM * sizeof(bf16));
  bf16*  WoT     = (bf16*) take((size_t)4 * DM * DM * sizeof(bf16));
  bf16*  W1T     = (bf16*) take((size_t)4 * DM * DI * sizeof(bf16));
  bf16*  W2T     = (bf16*) take((size_t)4 * DI * DM * sizeof(bf16));

  prefix_kernel<<<1, 32, 0, stream>>>(num_objs, offs);
  pad_kernel<<<M2, 256, 0, stream>>>(visual, text, num_objs, offs,
                                     vis_f, txt_f, vis_b, txt_b);

  for (int c = 0; c < 4; ++c) {
    size_t o512 = (size_t)c * DM * DM;
    size_t o2048 = (size_t)c * DM * DI;
    cvtw_kernel<<<(DM * DM + 255) / 256, 256, 0, stream>>>(Wq + o512, WqT + o512, DM, DM);
    cvtw_kernel<<<(DM * DM + 255) / 256, 256, 0, stream>>>(Wk + o512, WkT + o512, DM, DM);
    cvtw_kernel<<<(DM * DM + 255) / 256, 256, 0, stream>>>(Wv + o512, WvT + o512, DM, DM);
    cvtw_kernel<<<(DM * DM + 255) / 256, 256, 0, stream>>>(Wo + o512, WoT + o512, DM, DM);
    cvtw_kernel<<<(DM * DI + 255) / 256, 256, 0, stream>>>(W1 + o2048, W1T + o2048, DM, DI);
    cvtw_kernel<<<(DI * DM + 255) / 256, 256, 0, stream>>>(W2 + o2048, W2T + o2048, DI, DM);
  }

  dim3 g512(M2 / 64, DM / 128);    // (357, 4)
  dim3 g2048(M2 / 64, DI / 128);   // (357, 16)
  dim3 gattn(B_BATCH, 8);          // (240, 8)

  // Cell order: 1=SA_txt, 3=CA_txt, 0=SA_vis, 2=CA_vis
  const int order[4] = {1, 3, 0, 2};
  for (int ci = 0; ci < 4; ++ci) {
    int c = order[ci];
    bool q_is_vis = (c == 0 || c == 2);
    bool kv_is_vis = (c == 0 || c == 3);
    const bf16* qbuf = q_is_vis ? vis_b : txt_b;
    const float* qf  = q_is_vis ? vis_f : txt_f;
    const bf16* kvb  = kv_is_vis ? vis_b : txt_b;
    float* sum = q_is_vis ? sum_vis : sum_txt;
    int acc = ci & 1;   // second cell of each stream accumulates

    size_t o512 = (size_t)c * DM * DM;
    size_t o2048 = (size_t)c * DM * DI;
    const float* bq_c = bq + c * DM;   const float* bk_c = bk + c * DM;
    const float* bv_c = bv + c * DM;   const float* bo_c = bo + c * DM;
    const float* b1_c = b1 + c * DI;   const float* b2_c = b2 + c * DM;
    const float* g1_c = g1 + c * DM;   const float* be1_c = be1 + c * DM;
    const float* g2_c = g2 + c * DM;   const float* be2_c = be2 + c * DM;

    gemm_wmma<0><<<g512, 256, 0, stream>>>(qbuf, WqT + o512, bq_c, nullptr, nullptr, Qb, DM, DM);
    gemm_wmma<0><<<g512, 256, 0, stream>>>(kvb,  WkT + o512, bk_c, nullptr, nullptr, Kb, DM, DM);
    gemm_wmma<0><<<g512, 256, 0, stream>>>(kvb,  WvT + o512, bv_c, nullptr, nullptr, Vb, DM, DM);

    attn_wmma<<<gattn, 192, 0, stream>>>(Qb, Kb, Vb, Ctx, num_objs);

    gemm_wmma<2><<<g512, 256, 0, stream>>>(Ctx, WoT + o512, bo_c, qf, Tf, nullptr, DM, DM);
    ln_kernel<<<M2, 256, 0, stream>>>(Tf, g1_c, be1_c, num_objs, O1f, O1b, 0);

    gemm_wmma<1><<<g2048, 256, 0, stream>>>(O1b, W1T + o2048, b1_c, nullptr, nullptr, Hb, DI, DM);
    gemm_wmma<2><<<g512, 256, 0, stream>>>(Hb, W2T + o2048, b2_c, O1f, Tf, nullptr, DM, DI);
    ln_kernel<<<M2, 256, 0, stream>>>(Tf, g2_c, be2_c, num_objs, sum, nullptr, acc);
  }

  unpad_kernel<<<NTOK, 256, 0, stream>>>(sum_vis, sum_txt, offs, out);
}
